// DrugSpectral_35287451304635
// MI455X (gfx1250) — matrix operand order repaired and checked by reading, not compiled, not run on hardware
//
#include <hip/hip_runtime.h>

typedef __attribute__((ext_vector_type(16))) _Float16 v16h;
typedef __attribute__((ext_vector_type(8)))  float    v8f;

#define RT 4        // row tiles (of 16) per block  -> 64 rows/block
#define CT 6        // col tiles (96/16)
#define MAXCH 3     // max K chunks (96/32)

// ---------------------------------------------------------------------------
// Degree: deg[row[e]] += 1
// ---------------------------------------------------------------------------
__global__ void k_degree(const int* __restrict__ row, float* __restrict__ deg, int E) {
    int e = blockIdx.x * blockDim.x + threadIdx.x;
    if (e < E) atomicAdd(&deg[row[e]], 1.0f);
}

__global__ void k_dis(const float* __restrict__ deg, float* __restrict__ dis, int N) {
    int i = blockIdx.x * blockDim.x + threadIdx.x;
    if (i < N) {
        float d = deg[i];
        dis[i] = (d > 0.0f) ? rsqrtf(fmaxf(d, 1.0f)) : 0.0f;
    }
}

__global__ void k_edgew(const int* __restrict__ row, const int* __restrict__ col,
                        const float* __restrict__ dis, float* __restrict__ w, int E) {
    int e = blockIdx.x * blockDim.x + threadIdx.x;
    if (e < E) w[e] = -dis[row[e]] * dis[col[e]];
}

// ---------------------------------------------------------------------------
// WMMA GEMM:  Y[N,96] = A[N,Kdim] @ Wcat[Kdim,96], Wcat[k, k3*32+h] = W[k3,k,h]
// 192 threads = 6 waves; block owns 64 rows; wave w owns col-tile w.
// A / W staged into LDS in WMMA *fragment order*: each lane's 16 halves are
// contiguous -> operand loads are 2x ds_load_b128 into even-aligned VGPRs.
// ---------------------------------------------------------------------------
__global__ __launch_bounds__(192)
void k_gemm_wmma(const float* __restrict__ A, const float* __restrict__ W,
                 float* __restrict__ Y, int N, int Kdim, int nch) {
    __shared__ __align__(32) _Float16 Afrag[RT][MAXCH][32][16];  // 12 KB
    __shared__ __align__(32) _Float16 Bfrag[MAXCH][CT][32][16];  // 18 KB
    const int tid  = threadIdx.x;
    const int row0 = blockIdx.x * (RT * 16);

    // ---- stage A tiles in fragment order ----
    // A frag layout: lane l -> row M=l&15; elem j=(v,p): K = (v<4?0:16)+ (l>>4)*8 + 2*(v&3) + p
    const int nA = RT * nch * 512;
    for (int idx = tid; idx < nA; idx += 192) {
        int r    = idx / (nch * 512);
        int rem  = idx - r * (nch * 512);
        int cc   = rem / 512;
        int rem2 = rem - cc * 512;
        int l    = rem2 >> 4;
        int j    = rem2 & 15;
        int lh = l >> 4, lm = l & 15;
        int v = j >> 1, p = j & 1;
        int kb = ((v < 4) ? 0 : 16) + lh * 8 + 2 * (v & 3) + p;
        int k  = cc * 32 + kb;
        int gr = row0 + r * 16 + lm;
        float val = (gr < N && k < Kdim) ? A[(long)gr * Kdim + k] : 0.0f;
        Afrag[r][cc][l][j] = (_Float16)val;
    }
    // ---- stage W (B operand) in fragment order ----
    // B frag layout: lane l -> col N=l&15; elem j=(v,p): K = (v<4?0:16)+ (l>>4)*8 + 2*(v&3) + p
    const int nB = nch * CT * 512;
    for (int idx = tid; idx < nB; idx += 192) {
        int cc   = idx / (CT * 512);
        int rem  = idx - cc * (CT * 512);
        int t    = rem / 512;
        int rem2 = rem - t * 512;
        int l    = rem2 >> 4;
        int j    = rem2 & 15;
        int lh = l >> 4, lm = l & 15;
        int v = j >> 1, p = j & 1;
        int k = cc * 32 + ((v < 4) ? 0 : 16) + lh * 8 + 2 * (v & 3) + p;
        int c = t * 16 + lm;
        int k3 = c >> 5, h = c & 31;
        float val = (k < Kdim) ? W[(long)k3 * Kdim * 32 + (long)k * 32 + h] : 0.0f;
        Bfrag[cc][t][l][j] = (_Float16)val;
    }
    __syncthreads();

    const int wave = tid >> 5;     // 0..5 -> col tile
    const int lane = tid & 31;
    const int lh   = lane >> 4;
    const int lm   = lane & 15;
    const int c0   = wave * 16;
    const bool full = (row0 + RT * 16) <= N;

    for (int r = 0; r < RT; ++r) {
        v8f acc = {};
        for (int cc = 0; cc < nch; ++cc) {
            v16h a = *(const v16h*)&Afrag[r][cc][lane][0];
            v16h b = *(const v16h*)&Bfrag[cc][wave][lane][0];
            acc = __builtin_amdgcn_wmma_f32_16x16x32_f16(
                false, a, false, b, (short)0, acc, false, false);
        }
        // D: VGPR v -> row M = v + 8*laneHalf, col N = lm
        int rbase = row0 + r * 16 + lh * 8;
        if (full) {
            #pragma unroll
            for (int v = 0; v < 8; ++v)
                Y[(long)(rbase + v) * 96 + c0 + lm] = acc[v];
        } else {
            #pragma unroll
            for (int v = 0; v < 8; ++v)
                if (rbase + v < N) Y[(long)(rbase + v) * 96 + c0 + lm] = acc[v];
        }
    }
}

// ---------------------------------------------------------------------------
// prop: out[row[e], 0:32] += w[e] * in[col[e], 0:32]; 8 lanes per edge,
// float4 gathers + 4 f32 global atomics per lane.
// ---------------------------------------------------------------------------
__global__ void k_prop(const int* __restrict__ row, const int* __restrict__ col,
                       const float* __restrict__ w, const float* __restrict__ in,
                       int in_stride, float* __restrict__ out, int E) {
    long t = (long)blockIdx.x * blockDim.x + threadIdx.x;
    int e = (int)(t >> 3);
    int q = (int)(t & 7) * 4;
    if (e < E) {
        if (e + 16384 < E) {
            __builtin_prefetch(&col[e + 16384], 0, 0);   // global_prefetch_b8
            __builtin_prefetch(&row[e + 16384], 0, 0);
        }
        int c = col[e], r = row[e];
        float we = w[e];
        const float4 v4 = *(const float4*)&in[(long)c * in_stride + q];
        float* o = &out[(long)r * 32 + q];
        atomicAdd(o + 0, we * v4.x);
        atomicAdd(o + 1, we * v4.y);
        atomicAdd(o + 2, we * v4.z);
        atomicAdd(o + 3, we * v4.w);
    }
}

// T = Y[:,32:64] + 2*P
__global__ void k_mk_t(const float* __restrict__ Y, const float* __restrict__ P,
                       float* __restrict__ T, int N) {
    long t = (long)blockIdx.x * blockDim.x + threadIdx.x;
    int i = (int)(t >> 5), d = (int)(t & 31);
    if (i < N) T[(long)i * 32 + d] = Y[(long)i * 96 + 32 + d] + 2.0f * P[(long)i * 32 + d];
}

// H = relu(Y[:,0:32] - Y[:,64:96] + P + b)
__global__ void k_mk_h(const float* __restrict__ Y, const float* __restrict__ P,
                       const float* __restrict__ b, float* __restrict__ H, int N) {
    long t = (long)blockIdx.x * blockDim.x + threadIdx.x;
    int i = (int)(t >> 5), d = (int)(t & 31);
    if (i < N) {
        float v = Y[(long)i * 96 + d] - Y[(long)i * 96 + 64 + d]
                + P[(long)i * 32 + d] + b[d];
        H[(long)i * 32 + d] = fmaxf(v, 0.0f);
    }
}

// pooled[batch[i],d] += h[i,d]; cnt[batch[i]] += 1 (lane d==0)
__global__ void k_pool(const int* __restrict__ batch, const float* __restrict__ h,
                       float* __restrict__ pooled, float* __restrict__ cnt, int N) {
    long t = (long)blockIdx.x * blockDim.x + threadIdx.x;
    int i = (int)(t >> 5), d = (int)(t & 31);
    if (i < N) {
        int g = batch[i];
        atomicAdd(&pooled[(long)g * 32 + d], h[(long)i * 32 + d]);
        if (d == 0) atomicAdd(&cnt[g], 1.0f);
    }
}

// out[g] = (pooled[g]/max(cnt,1)) . Wfc + bfc
__global__ void k_final(const float* __restrict__ pooled, const float* __restrict__ cnt,
                        const float* __restrict__ Wfc, const float* __restrict__ bfc,
                        float* __restrict__ out, int G) {
    int g = blockIdx.x * blockDim.x + threadIdx.x;
    if (g < G) {
        float inv = 1.0f / fmaxf(cnt[g], 1.0f);
        float acc = bfc[0];
        #pragma unroll
        for (int d = 0; d < 32; ++d)
            acc += pooled[(long)g * 32 + d] * inv * Wfc[d];
        out[g] = acc;
    }
}

extern "C" void kernel_launch(void* const* d_in, const int* in_sizes, int n_in,
                              void* d_out, int out_size, void* d_ws, size_t ws_size,
                              hipStream_t stream) {
    const float* x   = (const float*)d_in[0];
    const float* W1  = (const float*)d_in[1];
    const float* b1  = (const float*)d_in[2];
    const float* W2  = (const float*)d_in[3];
    const float* b2  = (const float*)d_in[4];
    const float* Wfc = (const float*)d_in[5];
    const float* bfc = (const float*)d_in[6];
    const int*  edge = (const int*)d_in[7];
    const int*  batch= (const int*)d_in[8];

    const int N  = in_sizes[8];
    const int E  = in_sizes[7] / 2;
    const int IN = in_sizes[0] / N;       // 78
    const int G  = out_size;              // 2000
    const int* row = edge;
    const int* col = edge + E;

    // Workspace layout (floats)
    float* base   = (float*)d_ws;
    float* deg    = base;                 size_t off = (size_t)N;
    float* dis    = base + off;           off += (size_t)N;
    float* w      = base + off;           off += (size_t)E;
    float* Y      = base + off;           off += (size_t)N * 96;  // GEMM output (both layers)
    float* P      = base + off;           off += (size_t)N * 32;  // prop accumulator
    float* T      = base + off;           off += (size_t)N * 32;  // Tx intermediate / H2
    float* H1     = base + off;           off += (size_t)N * 32;  // layer-1 activation
    float* pooled = base + off;           off += (size_t)G * 32;
    float* cnt    = base + off;           off += (size_t)G;

    const int TB = 256;
    dim3 b256(TB);
    int gE   = (E + TB - 1) / TB;
    int gN   = (N + TB - 1) / TB;
    long nd  = (long)N * 32;
    int gND  = (int)((nd + TB - 1) / TB);
    long ed  = (long)E * 8;                       // 8 lanes per edge
    int gED  = (int)((ed + TB - 1) / TB);
    int gRow = (N + RT * 16 - 1) / (RT * 16);
    const int nch1 = (IN + 31) / 32;              // 3
    const int nch2 = 1;                           // K=32

    // ---- normalization ----
    hipMemsetAsync(deg, 0, (size_t)N * sizeof(float), stream);
    k_degree<<<gE, b256, 0, stream>>>(row, deg, E);
    k_dis<<<gN, b256, 0, stream>>>(deg, dis, N);
    k_edgew<<<gE, b256, 0, stream>>>(row, col, dis, w, E);

    // ---- layer 1: Y = X @ [W1_0|W1_1|W1_2]  (K=78 padded to 96) ----
    k_gemm_wmma<<<gRow, dim3(192), 0, stream>>>(x, W1, Y, N, IN, nch1);
    hipMemsetAsync(P, 0, (size_t)N * 32 * sizeof(float), stream);
    k_prop<<<gED, b256, 0, stream>>>(row, col, w, Y + 64, 96, P, E);   // P = L*(X@W1_2)
    k_mk_t<<<gND, b256, 0, stream>>>(Y, P, T, N);                      // T = X@W1_1 + 2P
    hipMemsetAsync(P, 0, (size_t)N * 32 * sizeof(float), stream);
    k_prop<<<gED, b256, 0, stream>>>(row, col, w, T, 32, P, E);        // P = L*T
    k_mk_h<<<gND, b256, 0, stream>>>(Y, P, b1, H1, N);                 // H1 = relu(...)

    // ---- layer 2: Y = H1 @ [W2_0|W2_1|W2_2]  (K=32) ----
    k_gemm_wmma<<<gRow, dim3(192), 0, stream>>>(H1, W2, Y, N, 32, nch2);
    hipMemsetAsync(P, 0, (size_t)N * 32 * sizeof(float), stream);
    k_prop<<<gED, b256, 0, stream>>>(row, col, w, Y + 64, 96, P, E);
    k_mk_t<<<gND, b256, 0, stream>>>(Y, P, T, N);
    hipMemsetAsync(P, 0, (size_t)N * 32 * sizeof(float), stream);
    k_prop<<<gED, b256, 0, stream>>>(row, col, w, T, 32, P, E);
    k_mk_h<<<gND, b256, 0, stream>>>(Y, P, b2, T, N);                  // T reused as H2

    // ---- pooling + FC ----
    hipMemsetAsync(pooled, 0, (size_t)G * 32 * sizeof(float), stream);
    hipMemsetAsync(cnt,    0, (size_t)G * sizeof(float), stream);
    k_pool<<<gND, b256, 0, stream>>>(batch, T, pooled, cnt, N);
    k_final<<<(G + TB - 1) / TB, b256, 0, stream>>>(pooled, cnt, Wfc, bfc, (float*)d_out, G);
}
